// Compress_10642928960020
// MI455X (gfx1250) — compile-verified, matrix operand order checked by reading, not run on hardware
//
#include <hip/hip_runtime.h>

typedef __attribute__((ext_vector_type(16))) __bf16 v16bf;
typedef __attribute__((ext_vector_type(8)))  float  v8f;
typedef __attribute__((ext_vector_type(4))) unsigned int v4u;
typedef __attribute__((ext_vector_type(8))) int v8i;
typedef __attribute__((ext_vector_type(4))) int v4i;

#define EPSF  1e-5f
#define PROPF 0.6981317007977318f  /* 40/180*pi */

// Halo-padded feature maps: [C][66][66], 1-pixel zero border -> branch-free im2col
#define HP   66
#define HPP  (66 * 66)   /* 4356 */
#define SROW 258         /* LDS k-row stride in ushorts: 256 data + 1 dword TDM pad */

#if defined(__HIP_DEVICE_COMPILE__) && __has_builtin(__builtin_amdgcn_tensor_load_to_lds) && __has_builtin(__builtin_amdgcn_s_wait_tensorcnt)
#define HAVE_TDM 1
#else
#define HAVE_TDM 0
#endif

__device__ __forceinline__ unsigned short f2bf(float f) {
  union { float f; unsigned u; } v; v.f = f;
  unsigned r = v.u + 0x7FFFu + ((v.u >> 16) & 1u);
  return (unsigned short)(r >> 16);
}
__device__ __forceinline__ float bf2f(unsigned short h) {
  union { float f; unsigned u; } v; v.u = ((unsigned)h) << 16;
  return v.f;
}

// bn(y + bias) = sc*y + sh
__global__ void k_affine(float* __restrict__ sc, float* __restrict__ sh,
                         const float* __restrict__ g, const float* __restrict__ bb,
                         const float* __restrict__ m, const float* __restrict__ v,
                         const float* __restrict__ bias, int C) {
  int c = blockIdx.x * blockDim.x + threadIdx.x;
  if (c < C) {
    float s = g[c] * rsqrtf(v[c] + EPSF);
    sc[c] = s;
    sh[c] = bb[c] + s * (bias[c] - m[c]);
  }
}

// 1x1 expand 32->192 + BN1 + ReLU, bf16 halo layout out
__global__ void __launch_bounds__(256) k_expand(
    const float* __restrict__ x, const float* __restrict__ w,
    const float* __restrict__ sc, const float* __restrict__ sh,
    unsigned short* __restrict__ h1) {
  __shared__ float wrow[32];
  int co = blockIdx.x, b = blockIdx.y, tid = threadIdx.x;
  if (tid < 32) wrow[tid] = w[co * 32 + tid];
  __syncthreads();
  float s = sc[co], t = sh[co];
  const float* xb = x + (size_t)b * 32 * 4096;
  unsigned short* o = h1 + ((size_t)b * 192 + co) * HPP;
  for (int p = tid; p < 4096; p += 256) {
    float acc = 0.f;
#pragma unroll
    for (int ci = 0; ci < 32; ++ci) acc += xb[ci * 4096 + p] * wrow[ci];
    o[((p >> 6) + 1) * HP + (p & 63) + 1] = f2bf(fmaxf(s * acc + t, 0.f));
  }
}

// depthwise 3x3 (pad 1 via halo) + GAP -> g[b][c]   (branch-free)
__global__ void __launch_bounds__(256) k_dwgap(
    const unsigned short* __restrict__ feat, const float* __restrict__ dw,
    float* __restrict__ g, int C) {
  __shared__ float red[256];
  __shared__ float wt[9];
  int c = blockIdx.x, b = blockIdx.y, tid = threadIdx.x;
  if (tid < 9) wt[tid] = dw[c * 9 + tid];
  __syncthreads();
  const unsigned short* f = feat + ((size_t)b * C + c) * HPP;
  float acc = 0.f;
  for (int p = tid; p < 4096; p += 256) {
    int ph = p >> 6, pw = p & 63;
    float s = 0.f;
#pragma unroll
    for (int tap = 0; tap < 9; ++tap)   // halo coords: (ph+tap/3, pw+tap%3)
      s += wt[tap] * bf2f(f[(ph + tap / 3) * HP + pw + tap % 3]);
    acc += s;
  }
  red[tid] = acc; __syncthreads();
  for (int st = 128; st > 0; st >>= 1) { if (tid < st) red[tid] += red[tid + st]; __syncthreads(); }
  if (tid == 0) g[b * C + c] = red[0] * (1.f / 4096.f);
}

// plain channel mean over [C][4096] (SE GAP)
__global__ void __launch_bounds__(256) k_cmean(
    const unsigned short* __restrict__ feat, float* __restrict__ g, int C) {
  __shared__ float red[256];
  int c = blockIdx.x, b = blockIdx.y, tid = threadIdx.x;
  const unsigned short* f = feat + ((size_t)b * C + c) * 4096;
  float acc = 0.f;
  for (int p = tid; p < 4096; p += 256) acc += bf2f(f[p]);
  red[tid] = acc; __syncthreads();
  for (int st = 128; st > 0; st >>= 1) { if (tid < st) red[tid] += red[tid + st]; __syncthreads(); }
  if (tid == 0) g[b * C + c] = red[0] * (1.f / 4096.f);
}

// LN -> ReLU -> fc -> alpha-scaled rotation matrices ra[b][n][81]
__global__ void __launch_bounds__(256) k_route(
    const float* __restrict__ g, const float* __restrict__ lnw,
    const float* __restrict__ lnb, const float* __restrict__ fcaw,
    const float* __restrict__ fcab, const float* __restrict__ fctw,
    float* __restrict__ ra, int C) {
  __shared__ float red[256];
  __shared__ float gn[192];
  __shared__ float al[4], th[4];
  int b = blockIdx.x, tid = threadIdx.x;
  float gv = (tid < C) ? g[b * C + tid] : 0.f;
  red[tid] = gv; __syncthreads();
  for (int st = 128; st > 0; st >>= 1) { if (tid < st) red[tid] += red[tid + st]; __syncthreads(); }
  float mu = red[0] / (float)C; __syncthreads();
  float d = (tid < C) ? (gv - mu) : 0.f;
  red[tid] = d * d; __syncthreads();
  for (int st = 128; st > 0; st >>= 1) { if (tid < st) red[tid] += red[tid + st]; __syncthreads(); }
  float inv = rsqrtf(red[0] / (float)C + EPSF);
  if (tid < C) gn[tid] = fmaxf(d * inv * lnw[tid] + lnb[tid], 0.f);
  __syncthreads();
  if (tid < 8) {
    int n = tid & 3;
    float acc = (tid < 4) ? fcab[n] : 0.f;
    const float* wrow = (tid < 4) ? (fcaw + n * C) : (fctw + n * C);
    for (int c = 0; c < C; ++c) acc += gn[c] * wrow[c];
    if (tid < 4) al[n] = 1.f / (1.f + expf(-acc));
    else        th[n] = acc / (1.f + fabsf(acc)) * PROPF;
  }
  __syncthreads();
  if (tid < 4) {
    float* Rp = ra + ((size_t)b * 4 + tid) * 81;
    float t = th[tid], A = al[tid];
    float x = cosf(t), y = sinf(t);
    float aa = x - y, bb = x * y, cc = x + y;
    float R[81];
#pragma unroll
    for (int i = 0; i < 81; ++i) R[i] = 0.f;
    if (t >= 0.f) {
      R[0]=aa; R[1]=1.f-aa;
      R[10]=x-bb; R[11]=bb; R[13]=1.f-cc+bb; R[14]=y-bb;
      R[20]=aa; R[23]=1.f-aa;
      R[27]=bb; R[28]=y-bb; R[30]=x-bb; R[31]=1.f-cc+bb;
      R[40]=1.f;
      R[49]=1.f-cc+bb; R[50]=x-bb; R[52]=y-bb; R[53]=bb;
      R[57]=1.f-aa; R[60]=aa;
      R[66]=y-bb; R[67]=1.f-cc+bb; R[69]=bb; R[70]=x-bb;
      R[79]=1.f-aa; R[80]=aa;
    } else {
      R[0]=cc; R[3]=1.f-cc;
      R[9]=-bb; R[10]=x+bb; R[12]=bb-y; R[13]=1.f-aa-bb;
      R[19]=1.f-cc; R[20]=cc;
      R[30]=x+bb; R[31]=1.f-aa-bb; R[33]=-bb; R[34]=bb-y;
      R[40]=1.f;
      R[46]=bb-y; R[47]=-bb; R[49]=1.f-aa-bb; R[50]=x+bb;
      R[60]=cc; R[61]=1.f-cc;
      R[67]=1.f-aa-bb; R[68]=bb-y; R[70]=x+bb; R[71]=-bb;
      R[77]=1.f-cc; R[80]=cc;
    }
#pragma unroll
    for (int i = 0; i < 81; ++i) Rp[i] = A * R[i];
  }
}

// per-sample weights wb[b][o][k], k = tap*Ci + ci (bf16)
__global__ void __launch_bounds__(256) k_buildwb(
    const float* __restrict__ W, const float* __restrict__ ra,
    unsigned short* __restrict__ wb, int Co, int Ci) {
  int K = Ci * 9;
  size_t total = (size_t)16 * Co * K;
  size_t e = (size_t)blockIdx.x * 256 + threadIdx.x;
  if (e >= total) return;
  int k = (int)(e % K); size_t bo = e / K;
  int o = (int)(bo % Co); int b = (int)(bo / Co);
  int i = k / Ci, ci = k % Ci;
  float acc = 0.f;
#pragma unroll
  for (int n = 0; n < 4; ++n) {
    const float* R  = ra + ((size_t)b * 4 + n) * 81 + i * 9;
    const float* Wp = W + (((size_t)n * Co + o) * Ci + ci) * 9;
#pragma unroll
    for (int j = 0; j < 9; ++j) acc += R[j] * Wp[j];
  }
  wb[e] = f2bf(acc);
}

// Implicit-im2col GEMM on WMMA bf16, B-tiles staged by the Tensor Data Mover.
// block = 256 thr (8 waves); tile 32(M) x 256(N); wave = 16x64 (4 wmma/K-step).
// X is halo [Ci][66][66] bf16; each K-chunk of 32 lies inside one 3x3 tap.
template<int M, int K, int Ci, bool HALO_OUT>
__global__ void __launch_bounds__(256) k_arcgemm(
    const unsigned short* __restrict__ A,   // [16][M][K] bf16
    const unsigned short* __restrict__ X,   // [16][Ci][66][66] bf16
    unsigned short* __restrict__ Y,         // halo or plain layout
    const float* __restrict__ sc, const float* __restrict__ sh) {
  __shared__ unsigned short Bs[2][32 * SROW];
  const int b = blockIdx.z;
  const int m_blk = blockIdx.x * 32;
  const int n_blk = blockIdx.y * 256;
  const int tid = threadIdx.x;
  const int lane = tid & 31, wave = tid >> 5;
  const int wm = (wave >> 2) << 4;   // 0 / 16
  const int wn = (wave & 3) << 6;    // 0,64,128,192
  const int kh = lane >> 4;          // K-half select (ISA 16-bit A/B layout)
  const int l16 = lane & 15;
  const int ph0 = n_blk >> 6;        // first image row of this 256-pixel tile
  constexpr int NSTEP = K / 32;

  v8f acc[4];
#pragma unroll
  for (int t = 0; t < 4; ++t) acc[t] = (v8f){0.f,0.f,0.f,0.f,0.f,0.f,0.f,0.f};

  const unsigned short* Arow = A + ((size_t)b * M + (m_blk + wm + l16)) * K;
  const size_t xbase = (size_t)b * Ci * HPP;

#if HAVE_TDM
  // Issue one TDM descriptor: 3-D tile 64(cols) x 4(rows,str 66) x 32(ch,str 4356),
  // 2-byte elements, packed into LDS with 1-dword pad per 128-dword (per-k) row.
  auto issue_tdm = [&](int s, int bufsel) {
    const int kk = s * 32;
    const int tap = kk / Ci;
    const int cib = kk - tap * Ci;
    const int dr = tap / 3 - 1, dc = tap % 3 - 1;
    unsigned long long ga = (unsigned long long)(uintptr_t)X +
        2ull * (xbase + (size_t)cib * HPP + (size_t)(ph0 + dr + 1) * HP + (size_t)(dc + 1));
    unsigned lds = (unsigned)(uintptr_t)(&Bs[bufsel][0]);  // low 32 bits = LDS offset
    v4u g0; v8i g1; v4i g2; v4i g3;
    g0.x = 1u;                                    // count=1, user descriptor
    g0.y = lds;                                   // lds_addr
    g0.z = (unsigned)ga;                          // global_addr[31:0]
    g0.w = (unsigned)((ga >> 32) & 0x1FFFFFFu) | (2u << 30);  // addr[56:32] | type=2
    g1[0] = (int)((1u << 16) | (1u << 20) | (6u << 22)); // data_size=2B, pad_en, interval=128dw, amount=1dw
    g1[1] = 0;                                    // tensor_dim0 = 1<<30 (low16 = 0)
    g1[2] = (int)0x4000u;                         // dim0 hi16 ; dim1 low16=0 (dim1=1<<30)
    g1[3] = (int)(0x4000u | (64u << 16));         // dim1 hi16 ; tile_dim0 = 64
    g1[4] = (int)(4u | (32u << 16));              // tile_dim1 = 4 rows, tile_dim2 = 32 ch
    g1[5] = HP;                                   // tensor_dim0_stride = 66
    g1[6] = (int)((unsigned)HPP << 16);           // stride0 hi=0 ; tensor_dim1_stride low16 = 4356
    g1[7] = 0;                                    // stride1[47:16] = 0
    g2[0] = (int)(1u << 30);                      // tensor_dim2 big (never OOB; halo guarantees)
    g2[1] = 0; g2[2] = 0; g2[3] = 0;
    g3[0] = 0; g3[1] = 0; g3[2] = 0; g3[3] = 0;
#if __clang_major__ >= 23
    v8i g4 = {0,0,0,0,0,0,0,0};
    __builtin_amdgcn_tensor_load_to_lds(g0, g1, g2, g3, g4, 0);
#else
    __builtin_amdgcn_tensor_load_to_lds(g0, g1, g2, g3, 0);
#endif
  };
#endif

  auto compute = [&](int s, const unsigned short* Bt) {
    const int kk = s * 32;
    union { v16bf v; unsigned u[8]; } af, bfr;
#pragma unroll
    for (int r = 0; r < 8; ++r) {   // A pairs at {0,2,4,6,16,18,20,22} + 8*khalf
      int ko = ((r < 4) ? (r << 1) : (16 + ((r - 4) << 1))) + (kh << 3);
      af.u[r] = *(const unsigned*)(Arow + kk + ko);
    }
#pragma unroll
    for (int t = 0; t < 4; ++t) {
      int n = wn + (t << 4) + l16;
#pragma unroll
      for (int r = 0; r < 8; ++r) {
        int ko = ((r < 4) ? (r << 1) : (16 + ((r - 4) << 1))) + (kh << 3);
        unsigned lo = Bt[ko * SROW + n];
        unsigned hi = Bt[(ko + 1) * SROW + n];
        bfr.u[r] = lo | (hi << 16);
      }
      acc[t] = __builtin_amdgcn_wmma_f32_16x16x32_bf16(false, af.v, false, bfr.v,
                                                       (short)0, acc[t], false, false);
    }
  };

#if HAVE_TDM
  if (wave == 0) issue_tdm(0, 0);
  for (int s = 0; s < NSTEP; ++s) {
    if (wave == 0) {
      if (s + 1 < NSTEP) {           // double buffer: prefetch next tile via TDM
        issue_tdm(s + 1, (s + 1) & 1);
        __builtin_amdgcn_s_wait_tensorcnt(1);   // oldest (tile s) complete
      } else {
        __builtin_amdgcn_s_wait_tensorcnt(0);
      }
    }
    __syncthreads();                  // tile s visible to all waves
    compute(s, &Bs[s & 1][0]);
    __syncthreads();                  // done reading before TDM s+2 overwrites
  }
#else
  // fallback: cooperative branch-free fill (halo removes all bounds checks)
  const unsigned short* Xb = X + xbase;
  for (int s = 0; s < NSTEP; ++s) {
    const int kk = s * 32;
    const int tap = kk / Ci, cib = kk - tap * Ci;
    __syncthreads();
#pragma unroll
    for (int t = 0; t < 32; ++t) {
      int e = tid + (t << 8);
      int kr = e >> 8, n = e & 255;
      int hr = ph0 + (n >> 6) + tap / 3, wc = (n & 63) + tap % 3;
      Bs[0][kr * SROW + n] = Xb[(size_t)(cib + kr) * HPP + hr * HP + wc];
    }
    __syncthreads();
    compute(s, &Bs[0][0]);
  }
#endif

  // epilogue: C/D layout VGPR r -> M = r (+8 for hi half-wave), N = lane%16
#pragma unroll
  for (int r = 0; r < 8; ++r) {
    int m = m_blk + wm + r + (kh << 3);
    float s0 = sc[m], t0 = sh[m];
#pragma unroll
    for (int t = 0; t < 4; ++t) {
      int n = n_blk + wn + (t << 4) + l16;
      float yv = fmaxf(s0 * acc[t][r] + t0, 0.f);
      size_t off;
      if (HALO_OUT) off = ((size_t)b * M + m) * HPP + (size_t)((n >> 6) + 1) * HP + (n & 63) + 1;
      else          off = ((size_t)b * M + m) * 4096 + (size_t)n;
      Y[off] = f2bf(yv);
    }
  }
}

// SE: fc(192->4)+ReLU -> fc(4->192)+sigmoid
__global__ void __launch_bounds__(256) k_se(
    const float* __restrict__ sgap, const float* __restrict__ w1,
    const float* __restrict__ b1, const float* __restrict__ w2,
    const float* __restrict__ b2, float* __restrict__ se) {
  __shared__ float s[192];
  int b = blockIdx.x, tid = threadIdx.x;
  if (tid < 192) s[tid] = sgap[b * 192 + tid];
  __syncthreads();
  if (tid < 192) {
    float h[4];
#pragma unroll
    for (int r = 0; r < 4; ++r) {
      float a = b1[r];
      for (int c = 0; c < 192; ++c) a += s[c] * w1[r * 192 + c];
      h[r] = fmaxf(a, 0.f);
    }
    float e = b2[tid];
#pragma unroll
    for (int r = 0; r < 4; ++r) e += h[r] * w2[tid * 4 + r];
    se[b * 192 + tid] = 1.f / (1.f + expf(-e));
  }
}

// SE scale + 1x1 proj (192->1) + bias + BN4, fp32 out
__global__ void __launch_bounds__(256) k_final(
    const unsigned short* __restrict__ h3, const float* __restrict__ se,
    const float* __restrict__ pw, const float* __restrict__ pb,
    const float* __restrict__ g4, const float* __restrict__ b4,
    const float* __restrict__ m4, const float* __restrict__ v4,
    float* __restrict__ out) {
  __shared__ float coef[192];
  int b = blockIdx.y, tid = threadIdx.x;
  if (tid < 192) coef[tid] = se[b * 192 + tid] * pw[tid];
  __syncthreads();
  int p = blockIdx.x * 256 + tid;
  const unsigned short* f = h3 + (size_t)b * 192 * 4096 + p;
  float acc = pb[0];
  for (int c = 0; c < 192; ++c) acc += bf2f(f[(size_t)c * 4096]) * coef[c];
  float s = g4[0] * rsqrtf(v4[0] + EPSF);
  out[(size_t)b * 4096 + p] = s * (acc - m4[0]) + b4[0];
}

extern "C" void kernel_launch(void* const* d_in, const int* in_sizes, int n_in,
                              void* d_out, int out_size, void* d_ws, size_t ws_size,
                              hipStream_t stream) {
  (void)in_sizes; (void)n_in; (void)out_size; (void)ws_size;
  const float* x        = (const float*)d_in[0];
  const float* expand_w = (const float*)d_in[1];
  const float* expand_b = (const float*)d_in[2];
  const float* bn1g = (const float*)d_in[3];  const float* bn1b = (const float*)d_in[4];
  const float* bn1m = (const float*)d_in[5];  const float* bn1v = (const float*)d_in[6];
  const float* arc1_w = (const float*)d_in[7];
  const float* arc1_b = (const float*)d_in[8];
  const float* r1_dw  = (const float*)d_in[9];
  const float* r1_lnw = (const float*)d_in[10]; const float* r1_lnb = (const float*)d_in[11];
  const float* r1_faw = (const float*)d_in[12]; const float* r1_fab = (const float*)d_in[13];
  const float* r1_ftw = (const float*)d_in[14];
  const float* bn2g = (const float*)d_in[15]; const float* bn2b = (const float*)d_in[16];
  const float* bn2m = (const float*)d_in[17]; const float* bn2v = (const float*)d_in[18];
  const float* arc2_w = (const float*)d_in[19];
  const float* arc2_b = (const float*)d_in[20];
  const float* r2_dw  = (const float*)d_in[21];
  const float* r2_lnw = (const float*)d_in[22]; const float* r2_lnb = (const float*)d_in[23];
  const float* r2_faw = (const float*)d_in[24]; const float* r2_fab = (const float*)d_in[25];
  const float* r2_ftw = (const float*)d_in[26];
  const float* bn3g = (const float*)d_in[27]; const float* bn3b = (const float*)d_in[28];
  const float* bn3m = (const float*)d_in[29]; const float* bn3v = (const float*)d_in[30];
  const float* se1_w = (const float*)d_in[31]; const float* se1_b = (const float*)d_in[32];
  const float* se2_w = (const float*)d_in[33]; const float* se2_b = (const float*)d_in[34];
  const float* proj_w = (const float*)d_in[35]; const float* proj_b = (const float*)d_in[36];
  const float* bn4g = (const float*)d_in[37]; const float* bn4b = (const float*)d_in[38];
  const float* bn4m = (const float*)d_in[39]; const float* bn4v = (const float*)d_in[40];
  float* out = (float*)d_out;

  char* ws = (char*)d_ws;
  size_t off = 0;
  auto alloc = [&](size_t bytes) -> void* {
    void* p = (void*)(ws + off);
    off = (off + bytes + 255) & ~(size_t)255;
    return p;
  };
  const size_t h1_bytes = (size_t)16 * 192 * HPP * 2;
  const size_t h2_bytes = (size_t)16 * 96 * HPP * 2;
  unsigned short* h1  = (unsigned short*)alloc(h1_bytes);           // halo bf16
  unsigned short* h2  = (unsigned short*)alloc(h2_bytes);           // halo bf16
  unsigned short* h3  = (unsigned short*)alloc((size_t)16*192*4096*2); // plain bf16
  unsigned short* wb1 = (unsigned short*)alloc((size_t)16*96*1728*2);
  unsigned short* wb2 = (unsigned short*)alloc((size_t)16*192*864*2);
  float* g1   = (float*)alloc((size_t)16*192*4);
  float* g2   = (float*)alloc((size_t)16*96*4);
  float* ra1  = (float*)alloc((size_t)16*4*81*4);
  float* ra2  = (float*)alloc((size_t)16*4*81*4);
  float* sgap = (float*)alloc((size_t)16*192*4);
  float* sev  = (float*)alloc((size_t)16*192*4);
  float* sc1 = (float*)alloc(192*4); float* sh1 = (float*)alloc(192*4);
  float* sc2 = (float*)alloc(96*4);  float* sh2 = (float*)alloc(96*4);
  float* sc3 = (float*)alloc(192*4); float* sh3 = (float*)alloc(192*4);

  // zero halo buffers (borders must read as zero-padding)
  hipMemsetAsync(h1, 0, h1_bytes, stream);
  hipMemsetAsync(h2, 0, h2_bytes, stream);

  k_affine<<<1, 192, 0, stream>>>(sc1, sh1, bn1g, bn1b, bn1m, bn1v, expand_b, 192);
  k_affine<<<1, 96,  0, stream>>>(sc2, sh2, bn2g, bn2b, bn2m, bn2v, arc1_b, 96);
  k_affine<<<1, 192, 0, stream>>>(sc3, sh3, bn3g, bn3b, bn3m, bn3v, arc2_b, 192);

  k_expand<<<dim3(192, 16), 256, 0, stream>>>(x, expand_w, sc1, sh1, h1);

  // arc1: 16 x GEMM(96 x 1728 x 4096) on WMMA bf16, TDM-staged im2col
  k_dwgap<<<dim3(192, 16), 256, 0, stream>>>(h1, r1_dw, g1, 192);
  k_route<<<16, 256, 0, stream>>>(g1, r1_lnw, r1_lnb, r1_faw, r1_fab, r1_ftw, ra1, 192);
  k_buildwb<<<(16*96*1728 + 255)/256, 256, 0, stream>>>(arc1_w, ra1, wb1, 96, 192);
  k_arcgemm<96, 1728, 192, true><<<dim3(3, 16, 16), 256, 0, stream>>>(wb1, h1, h2, sc2, sh2);

  // arc2: 16 x GEMM(192 x 864 x 4096)
  k_dwgap<<<dim3(96, 16), 256, 0, stream>>>(h2, r2_dw, g2, 96);
  k_route<<<16, 256, 0, stream>>>(g2, r2_lnw, r2_lnb, r2_faw, r2_fab, r2_ftw, ra2, 96);
  k_buildwb<<<(16*192*864 + 255)/256, 256, 0, stream>>>(arc2_w, ra2, wb2, 192, 96);
  k_arcgemm<192, 864, 96, false><<<dim3(6, 16, 16), 256, 0, stream>>>(wb2, h2, h3, sc3, sh3);

  // SE + projection
  k_cmean<<<dim3(192, 16), 256, 0, stream>>>(h3, sgap, 192);
  k_se<<<16, 256, 0, stream>>>(sgap, se1_w, se1_b, se2_w, se2_b, sev);
  k_final<<<dim3(16, 16), 256, 0, stream>>>(h3, sev, proj_w, proj_b,
                                            bn4g, bn4b, bn4m, bn4v, out);
}
